// GlobalTransform_45603962749431
// MI455X (gfx1250) — matrix operand order repaired and checked by reading, not compiled, not run on hardware
//
#include <hip/hip_runtime.h>

// ---------------------------------------------------------------------------
// PointNet T-Net fused pipeline for gfx1250 (wave32, WMMA 16x16x32 f16->f32)
//   K1: fused 4-layer MLP (3->64->128->256->64), activations in LDS,
//       GEMMs on v_wmma_f32_16x16x32_f16, h4 spilled fp32 to ws.
//   K2: per-(b,c) stable softmax over 8192 points -> 8x8 transform (+I).
//   K3: out[b,m,k] = sum_l T[b,k,l] * x[b,m,l]   (memory-bound).
// ---------------------------------------------------------------------------

typedef _Float16 v16h  __attribute__((ext_vector_type(16)));
typedef _Float16 half8 __attribute__((ext_vector_type(8)));
typedef float    v8f   __attribute__((ext_vector_type(8)));

#define NPTS    8192
#define NBATCH  16
#define MPTS    8192
#define LDIM    8
#define TILE_P  64
#define STRIDE0 264   // buf0 channel stride (max 256 ch + 8 pad), mult of 8
#define STRIDE1 136   // buf1 channel stride (max 128 ch + 8 pad), mult of 8

// f16 weight offsets inside ws (elements)
#define W2_OFF  0        // 128*64  = 8192
#define W3_OFF  8192     // 256*128 = 32768
#define W4_OFF  40960    // 64*256  = 16384
#define WTOTAL  57344
// folded BN scale/bias (f32) layout inside st[]
#define S1_OFF 0
#define T1_OFF 64
#define S2_OFF 128
#define T2_OFF 256
#define S3_OFF 384
#define T3_OFF 640
#define S4_OFF 896
#define T4_OFF 960

__device__ __forceinline__ v16h cat8(half8 lo, half8 hi) {
  return __builtin_shufflevector(lo, hi, 0,1,2,3,4,5,6,7,8,9,10,11,12,13,14,15);
}

// ---------------------------------------------------------------------------
// Prep: convert W2/W3/W4 fp32 -> f16 in ws, fold BN into per-channel (s, t):
//   y = s * (W.x) + t,  s = gamma * rsqrt(var+eps),  t = s*(b-mean) + beta
// ---------------------------------------------------------------------------
struct PrepParams {
  const float* W2; const float* W3; const float* W4;
  const float* bb[4]; const float* ga[4]; const float* be[4];
  const float* mu[4]; const float* va[4];
};

__global__ __launch_bounds__(256) void prep_kernel(PrepParams p,
                                                   _Float16* __restrict__ wout,
                                                   float* __restrict__ st) {
  int gid = blockIdx.x * 256 + threadIdx.x;
  if (gid < 8192) {
    wout[W2_OFF + gid] = (_Float16)p.W2[gid];
  } else if (gid < 40960) {
    wout[gid] = (_Float16)p.W3[gid - 8192];
  } else if (gid < WTOTAL) {
    wout[gid] = (_Float16)p.W4[gid - 40960];
  } else if (gid < WTOTAL + 512) {
    int idx = gid - WTOTAL;
    int l, c, sOff, cout;
    if      (idx < 64)  { l = 0; c = idx;       sOff = S1_OFF; cout = 64;  }
    else if (idx < 192) { l = 1; c = idx - 64;  sOff = S2_OFF; cout = 128; }
    else if (idx < 448) { l = 2; c = idx - 192; sOff = S3_OFF; cout = 256; }
    else                { l = 3; c = idx - 448; sOff = S4_OFF; cout = 64;  }
    float inv = rsqrtf(p.va[l][c] + 1e-5f);
    float s = p.ga[l][c] * inv;
    float t = (p.bb[l][c] - p.mu[l][c]) * s + p.be[l][c];
    st[sOff + c] = s;
    st[sOff + cout + c] = t;
  }
}

// ---------------------------------------------------------------------------
// One hidden GEMM layer on WMMA: out[cout,64pts] = relu(s*(W@in)+t), f16 LDS
// A frag (16x32 f16): lane m=lane&15, g=lane>>4; K chunks [8g..8g+7],[16+8g..]
// B frag (32x16 f16): lane n=lane&15, g=lane>>4; K chunks [8g..8g+7],[16+8g..]
// D (16x16 f32): vgpr r -> row 8g+r, col lane&15
// ---------------------------------------------------------------------------
__device__ __forceinline__ void gemm_layer(
    const _Float16* __restrict__ Wf, int cin, int cout,
    const float* __restrict__ s, const float* __restrict__ t,
    const _Float16* in, int inStride, _Float16* out, int outStride,
    int wave, int lane)
{
  const int idx = lane & 15, g = (lane >> 4) & 1;
  const int ntiles = (cout >> 4) << 2;   // (cout/16) cblocks x 4 pblocks
  for (int tile = wave; tile < ntiles; tile += 8) {
    const int cb = tile >> 2, pb = tile & 3;
    v8f acc = {};
    const _Float16* wrow = Wf + (cb * 16 + idx) * cin;
    const _Float16* brow = in + (pb * 16 + idx) * inStride;
    for (int kb = 0; kb < cin; kb += 32) {
      half8 a0 = *(const half8*)(wrow + kb + 8 * g);
      half8 a1 = *(const half8*)(wrow + kb + 16 + 8 * g);
      half8 b0 = *(const half8*)(brow + kb + 8 * g);
      half8 b1 = *(const half8*)(brow + kb + 16 + 8 * g);
      acc = __builtin_amdgcn_wmma_f32_16x16x32_f16(
          false, cat8(a0, a1), false, cat8(b0, b1), (short)0, acc, false, false);
    }
    const int cbase = cb * 16 + 8 * g;
    half8 o;
#pragma unroll
    for (int r = 0; r < 8; ++r) {
      float v = acc[r] * s[cbase + r] + t[cbase + r];
      o[r] = (_Float16)fmaxf(v, 0.0f);
    }
    *(half8*)(out + (pb * 16 + idx) * outStride + cbase) = o;
  }
}

// Last layer: same GEMM, but epilogue writes fp32 h4[b][c][n] to global ws.
__device__ __forceinline__ void final_layer(
    const _Float16* __restrict__ Wf, int cin,
    const float* __restrict__ s, const float* __restrict__ t,
    const _Float16* in, int inStride,
    float* __restrict__ h4, int b, int base_n, int wave, int lane)
{
  const int idx = lane & 15, g = (lane >> 4) & 1;
  for (int tile = wave; tile < 16; tile += 8) {   // 4 cblocks x 4 pblocks
    const int cb = tile >> 2, pb = tile & 3;
    v8f acc = {};
    const _Float16* wrow = Wf + (cb * 16 + idx) * cin;
    const _Float16* brow = in + (pb * 16 + idx) * inStride;
    for (int kb = 0; kb < cin; kb += 32) {
      half8 a0 = *(const half8*)(wrow + kb + 8 * g);
      half8 a1 = *(const half8*)(wrow + kb + 16 + 8 * g);
      half8 b0 = *(const half8*)(brow + kb + 8 * g);
      half8 b1 = *(const half8*)(brow + kb + 16 + 8 * g);
      acc = __builtin_amdgcn_wmma_f32_16x16x32_f16(
          false, cat8(a0, a1), false, cat8(b0, b1), (short)0, acc, false, false);
    }
    const int cbase = cb * 16 + 8 * g;
    const int n = base_n + pb * 16 + idx;
#pragma unroll
    for (int r = 0; r < 8; ++r) {
      const int c = cbase + r;
      float v = fmaxf(acc[r] * s[c] + t[c], 0.0f);
      h4[((size_t)(b * 64 + c)) * NPTS + n] = v;
    }
  }
}

// ---------------------------------------------------------------------------
// K1: fused MLP. One block = one batch x 64-point tile. 256 thr = 8 waves.
// ---------------------------------------------------------------------------
__global__ __launch_bounds__(256) void mlp_kernel(
    const float* __restrict__ partial, const float* __restrict__ W1,
    const _Float16* __restrict__ wf16, const float* __restrict__ st,
    float* __restrict__ h4)
{
  __shared__ _Float16 buf0[TILE_P * STRIDE0];  // h1 (64ch) then h3 (256ch)
  __shared__ _Float16 buf1[TILE_P * STRIDE1];  // h2 (128ch)
  __shared__ float    pstage[3 * TILE_P];

  const int b = blockIdx.x >> 7;            // / (8192/64)
  const int base_n = (blockIdx.x & 127) * TILE_P;
  const int tid = threadIdx.x;
  const int wave = tid >> 5, lane = tid & 31;

  // stage partial[b, 0..2, base_n..base_n+63]
  if (tid < 192) {
    int ch = tid >> 6, p = tid & 63;
    pstage[ch * TILE_P + p] = partial[((size_t)b * 3 + ch) * NPTS + base_n + p];
  }
  __syncthreads();

  // Layer 1 (3 -> 64), plain VALU, write f16 point-major into buf0
  {
    const int c = tid & 63, pblk = tid >> 6;
    float w0 = W1[c * 3 + 0], w1 = W1[c * 3 + 1], w2 = W1[c * 3 + 2];
    float sc = st[S1_OFF + c], tc = st[T1_OFF + c];
#pragma unroll
    for (int i = 0; i < 16; ++i) {
      int p = pblk * 16 + i;
      float v = sc * (w0 * pstage[p] + w1 * pstage[TILE_P + p] +
                      w2 * pstage[2 * TILE_P + p]) + tc;
      buf0[p * STRIDE0 + c] = (_Float16)fmaxf(v, 0.0f);
    }
  }
  __syncthreads();

  // Layer 2 (64 -> 128): buf0 -> buf1
  gemm_layer(wf16 + W2_OFF, 64, 128, st + S2_OFF, st + T2_OFF,
             buf0, STRIDE0, buf1, STRIDE1, wave, lane);
  __syncthreads();

  // Layer 3 (128 -> 256): buf1 -> buf0
  gemm_layer(wf16 + W3_OFF, 128, 256, st + S3_OFF, st + T3_OFF,
             buf1, STRIDE1, buf0, STRIDE0, wave, lane);
  __syncthreads();

  // Layer 4 (256 -> 64): buf0 -> global fp32 h4
  final_layer(wf16 + W4_OFF, 256, st + S4_OFF, st + T4_OFF,
              buf0, STRIDE0, h4, b, base_n, wave, lane);
}

// ---------------------------------------------------------------------------
// K2: per (b,c) row: stable softmax over N points, weighted sum -> T[b][c]
// ---------------------------------------------------------------------------
__global__ __launch_bounds__(256) void softmax_transform_kernel(
    const float* __restrict__ h4, float* __restrict__ T)
{
  __shared__ float red[256];
  const int row = blockIdx.x;                 // b*64 + c
  const float* hr = h4 + (size_t)row * NPTS;
  const int tid = threadIdx.x;

  float m = -1e30f;
  for (int i = tid; i < NPTS; i += 256) m = fmaxf(m, hr[i]);
  red[tid] = m; __syncthreads();
  for (int s = 128; s > 0; s >>= 1) {
    if (tid < s) red[tid] = fmaxf(red[tid], red[tid + s]);
    __syncthreads();
  }
  m = red[0]; __syncthreads();

  float se = 0.0f, sw = 0.0f;
  for (int i = tid; i < NPTS; i += 256) {
    float h = hr[i];
    float e = __expf(h - m);
    se += e; sw += e * h;
  }
  red[tid] = se; __syncthreads();
  for (int s = 128; s > 0; s >>= 1) {
    if (tid < s) red[tid] += red[tid + s];
    __syncthreads();
  }
  se = red[0]; __syncthreads();
  red[tid] = sw; __syncthreads();
  for (int s = 128; s > 0; s >>= 1) {
    if (tid < s) red[tid] += red[tid + s];
    __syncthreads();
  }
  sw = red[0];

  if (tid == 0) {
    int c = row & 63, k = c >> 3, l = c & 7;
    T[row] = sw / se + (k == l ? 1.0f : 0.0f);
  }
}

// ---------------------------------------------------------------------------
// K3: out[b, m*8+k] = sum_l T[b][k*8+l] * x[b][m*8+l]
// ---------------------------------------------------------------------------
__global__ __launch_bounds__(256) void apply_kernel(
    const float* __restrict__ x, const float* __restrict__ T,
    float* __restrict__ out)
{
  __shared__ float Ts[64];
  const int b  = blockIdx.x >> 5;       // / (8192/256)
  const int mb = blockIdx.x & 31;
  if (threadIdx.x < 64) Ts[threadIdx.x] = T[b * 64 + threadIdx.x];
  __syncthreads();

  const int m = mb * 256 + threadIdx.x;
  const size_t base = (size_t)b * (MPTS * LDIM) + (size_t)m * LDIM;
  const float4 x0 = *(const float4*)(x + base);
  const float4 x1 = *(const float4*)(x + base + 4);
  const float xl[8] = {x0.x, x0.y, x0.z, x0.w, x1.x, x1.y, x1.z, x1.w};
  float o[8];
#pragma unroll
  for (int k = 0; k < 8; ++k) {
    float acc = 0.0f;
#pragma unroll
    for (int l = 0; l < 8; ++l) acc += Ts[k * 8 + l] * xl[l];
    o[k] = acc;
  }
  *(float4*)(out + base)     = make_float4(o[0], o[1], o[2], o[3]);
  *(float4*)(out + base + 4) = make_float4(o[4], o[5], o[6], o[7]);
}

// ---------------------------------------------------------------------------
extern "C" void kernel_launch(void* const* d_in, const int* in_sizes, int n_in,
                              void* d_out, int out_size, void* d_ws, size_t ws_size,
                              hipStream_t stream) {
  (void)in_sizes; (void)n_in; (void)out_size; (void)ws_size;
  const float* partial = (const float*)d_in[0];
  const float* x       = (const float*)d_in[1];
  // params flattened: per layer W, b, gamma, beta, mean, var
  const float* W1 = (const float*)d_in[2];

  PrepParams pp;
  pp.W2 = (const float*)d_in[8];
  pp.W3 = (const float*)d_in[14];
  pp.W4 = (const float*)d_in[20];
  for (int l = 0; l < 4; ++l) {
    int base = 2 + 6 * l;
    pp.bb[l] = (const float*)d_in[base + 1];
    pp.ga[l] = (const float*)d_in[base + 2];
    pp.be[l] = (const float*)d_in[base + 3];
    pp.mu[l] = (const float*)d_in[base + 4];
    pp.va[l] = (const float*)d_in[base + 5];
  }

  // workspace layout
  char* wsb = (char*)d_ws;
  _Float16* wf16 = (_Float16*)wsb;                       // 114688 B
  float*    st   = (float*)(wsb + 114688);               //   4096 B
  float*    h4   = (float*)(wsb + 131072);               // 33.55 MB
  float*    T    = (float*)(wsb + 131072 + (size_t)NBATCH * 64 * NPTS * 4);

  float* out = (float*)d_out;

  prep_kernel<<<(WTOTAL + 512 + 255) / 256, 256, 0, stream>>>(pp, wf16, st);
  mlp_kernel<<<NBATCH * (NPTS / TILE_P), 256, 0, stream>>>(partial, W1, wf16, st, h4);
  softmax_transform_kernel<<<NBATCH * 64, 256, 0, stream>>>(h4, T);
  apply_kernel<<<NBATCH * (MPTS / 256), 256, 0, stream>>>(x, T, out);
}
